// FCLSTM_62612033241449
// MI455X (gfx1250) — matrix operand order repaired and checked by reading, not compiled, not run on hardware
//
#include <hip/hip_runtime.h>
#include <hip/hip_bf16.h>
#include <stdint.h>

#define BATCH 256
#define NPTS  300
#define TLEN  64
#define FIN   3
#define FOUT  2
#define UDIM  256
#define GDIM  1024   // 4*UDIM
#define KIN   900    // NPTS*FIN
#define KINP  928    // padded to multiple of 32
#define NDOUT 600    // NPTS*FOUT
#define NDOUTP 640   // padded to multiple of 64 (branch-free GEMM tiling)
#define BT   (BATCH*TLEN)

typedef __attribute__((ext_vector_type(16))) __bf16        bf16x16;
typedef __attribute__((ext_vector_type(8)))  float         floatx8;
typedef __attribute__((ext_vector_type(8)))  unsigned int  uintx8;

__device__ __forceinline__ uint16_t f2bf(float f) {
  uint32_t u = __float_as_uint(f);
  uint32_t r = u + 0x7FFFu + ((u >> 16) & 1u);   // round-to-nearest-even
  return (uint16_t)(r >> 16);
}

// A fragment: 16x32 bf16, row-major source, leading dim lda (elements).
// ISA layout: lanes 0-15 (M=lane): V0-3 -> K=0..7, V4-7 -> K=16..23;
// lanes 16-31: +8 on K. Each half is 16 contiguous bytes -> b128 loads.
__device__ __forceinline__ bf16x16 load_a_frag(const uint16_t* A, int lda, int kb) {
  unsigned l = threadIdx.x & 31u;
  int m = l & 15, hi = l >> 4;
  const uint16_t* base = A + (size_t)m * lda + kb + hi * 8;
  uint4 v0 = *(const uint4*)(base);
  uint4 v1 = *(const uint4*)(base + 16);
  uintx8 u;
  u[0] = v0.x; u[1] = v0.y; u[2] = v0.z; u[3] = v0.w;
  u[4] = v1.x; u[5] = v1.y; u[6] = v1.z; u[7] = v1.w;
  return __builtin_bit_cast(bf16x16, u);
}

// B fragment: 32x16 bf16 from TRANSPOSED weights Bt (N rows, K cols, ld=ldb).
// ISA layout: lanes 0-15 (N=lane): V0-3 -> K=0..7, V4-7 -> K=16..23;
// lanes 16-31: +8 on K. Each half is 16 contiguous bytes -> b128 loads.
__device__ __forceinline__ bf16x16 load_b_frag(const uint16_t* Bt, int ldb, int kb) {
  unsigned l = threadIdx.x & 31u;
  int n = l & 15, hi = l >> 4;
  const uint16_t* base = Bt + (size_t)n * ldb + kb + hi * 8;
  uint4 v0 = *(const uint4*)(base);
  uint4 v1 = *(const uint4*)(base + 16);
  uintx8 u;
  u[0] = v0.x; u[1] = v0.y; u[2] = v0.z; u[3] = v0.w;
  u[4] = v1.x; u[5] = v1.y; u[6] = v1.z; u[7] = v1.w;
  return __builtin_bit_cast(bf16x16, u);
}

__device__ __forceinline__ floatx8 wmma_bf16(bf16x16 a, bf16x16 b, floatx8 c) {
  return __builtin_amdgcn_wmma_f32_16x16x32_bf16(false, a, false, b, (short)0, c,
                                                 false, false);
}

// ---- pack x (B,N,T,F) -> XA[(t*B+b), KINP] bf16, zero-padded K ----
__global__ void __launch_bounds__(256) prep_xa_k(const float* __restrict__ x,
                                                 uint16_t* __restrict__ XA) {
  size_t i = (size_t)blockIdx.x * 256 + threadIdx.x;
  const size_t total = (size_t)BT * KINP;
  if (i >= total) return;
  int c = (int)(i % KINP);
  size_t r = i / KINP;
  int b = (int)(r % BATCH);
  int t = (int)(r / BATCH);
  uint16_t v = 0;
  if (c < KIN) {
    int nn = c / FIN, f = c - nn * FIN;
    v = f2bf(x[(((size_t)b * NPTS + nn) * TLEN + t) * FIN + f]);
  }
  XA[i] = v;
}

// ---- convert f32 (K,N) weights -> bf16 transposed (Npad, Kpad), zero pad ----
__global__ void __launch_bounds__(256) convert_wt_k(const float* __restrict__ S,
                                                    uint16_t* __restrict__ D,
                                                    int K, int N, int Kpad, int Npad) {
  size_t i = (size_t)blockIdx.x * 256 + threadIdx.x;
  size_t total = (size_t)Npad * Kpad;
  if (i >= total) return;
  int k  = (int)(i % Kpad);
  int nn = (int)(i / Kpad);
  float v = (k < K && nn < N) ? S[(size_t)k * N + nn] : 0.f;
  D[i] = f2bf(v);
}

// ---- branch-free, software-pipelined WMMA GEMM (NT=4, ~120 live VGPRs) ----
// C(M,Npad) f32 = A(M,Kpad)bf16 @ Bt(Npad,Kpad)bf16 + bias
// block = 8 waves = 128 rows x 64 cols; wave = 16x64.
// grid = (M/128, Npad/64). Next k-step's fragments prefetched into fresh
// registers (unroll-2 rotation) so waits overlap the 4 WMMAs.
template <int NT>
__global__ void __launch_bounds__(256) gemm_bf16(
    const uint16_t* __restrict__ A, int lda,
    const uint16_t* __restrict__ Bt, int ldb,
    const float* __restrict__ bias, int nbias,
    float* __restrict__ C, int ldc, int Kpad) {
  int wave = threadIdx.x >> 5;
  int row0 = blockIdx.x * 128 + wave * 16;
  int n0   = blockIdx.y * (16 * NT);
  unsigned l = threadIdx.x & 31u;
  int n = l & 15, hi = l >> 4;

  const uint16_t* Arow = A + (size_t)row0 * lda;
  const uint16_t* Bp[NT];
#pragma unroll
  for (int i = 0; i < NT; ++i) Bp[i] = Bt + (size_t)(n0 + 16 * i) * ldb;

  floatx8 acc[NT] = {};
  bf16x16 a = load_a_frag(Arow, lda, 0);
  bf16x16 bf[NT];
#pragma unroll
  for (int i = 0; i < NT; ++i) bf[i] = load_b_frag(Bp[i], ldb, 0);

#pragma unroll 2
  for (int kb = 0; kb < Kpad; kb += 32) {
    int kn = (kb + 32 < Kpad) ? kb + 32 : 0;   // last iter: harmless re-read
    bf16x16 an = load_a_frag(Arow, lda, kn);
    bf16x16 bn[NT];
#pragma unroll
    for (int i = 0; i < NT; ++i) bn[i] = load_b_frag(Bp[i], ldb, kn);
#pragma unroll
    for (int i = 0; i < NT; ++i) acc[i] = wmma_bf16(a, bf[i], acc[i]);
    a = an;
#pragma unroll
    for (int i = 0; i < NT; ++i) bf[i] = bn[i];
  }

#pragma unroll
  for (int tIdx = 0; tIdx < NT; ++tIdx) {
    int col = n0 + tIdx * 16 + n;
    int cc  = col < nbias ? col : (nbias - 1);
    float bv = bias[cc] * (col < nbias ? 1.f : 0.f);   // lane-local mask
    float* Cp = C + (size_t)row0 * ldc + col;
#pragma unroll
    for (int j = 0; j < 8; ++j) Cp[(size_t)(j + 8 * hi) * ldc] = acc[tIdx][j] + bv;
  }
}

// ---- LSTM recurrence: 1 block = 16 batch rows, 16 waves (4/SIMD), h in LDS.
// Wave w owns hidden column block w (16 cols) and its 4 gate tiles; U streams
// from L2 with a software-pipelined k-loop (prefetch next A-from-LDS and
// 4 B-from-global while 4 WMMAs consume the current fragments).
// ~140 live VGPRs -> no spills; 4 waves/SIMD hide the L2 latency bubbles.
__global__ void __launch_bounds__(512) lstm_rec(
    const float* __restrict__ ZX,      // (T*B, GDIM) f32, rows t*B+b
    const uint16_t* __restrict__ Ut,   // (GDIM, UDIM) bf16 transposed
    uint16_t* __restrict__ Hout) {     // (T*B, UDIM) bf16
  __shared__ uint16_t hbuf[16 * UDIM];
  int tid  = threadIdx.x;
  int nblk = tid >> 5;                 // wave id = hidden column block (0..15)
  unsigned l = tid & 31u;
  int n = l & 15, hi = l >> 4;
  int b0 = blockIdx.x * 16;

  unsigned* hz = (unsigned*)hbuf;
  for (int i = tid; i < 16 * UDIM / 2; i += 512) hz[i] = 0u;
  __syncthreads();

  // B row panels for this wave's 4 gate tiles (g*256 + nblk*16)
  const uint16_t* Bp[4];
#pragma unroll
  for (int g = 0; g < 4; ++g)
    Bp[g] = Ut + (size_t)(g * UDIM + nblk * 16) * UDIM;

  floatx8 cstate = {};

  for (int t = 0; t < TLEN; ++t) {
    // seed accumulators with precomputed zx for this step
    floatx8 acc[4];
#pragma unroll
    for (int g = 0; g < 4; ++g) {
      int col = g * UDIM + nblk * 16 + n;
      const float* zp = ZX + ((size_t)t * BATCH + b0) * GDIM + col;
      floatx8 v;
#pragma unroll
      for (int j = 0; j < 8; ++j) v[j] = zp[(size_t)(j + 8 * hi) * GDIM];
      acc[g] = v;
    }

    // z += h_{t-1} @ U : software-pipelined over 8 k-steps
    bf16x16 a = load_a_frag(hbuf, UDIM, 0);
    bf16x16 bf[4];
#pragma unroll
    for (int g = 0; g < 4; ++g) bf[g] = load_b_frag(Bp[g], UDIM, 0);

#pragma unroll 2
    for (int kb = 0; kb < UDIM; kb += 32) {
      int kn = (kb + 32 < UDIM) ? kb + 32 : 0;  // last iter: harmless re-read
      bf16x16 an = load_a_frag(hbuf, UDIM, kn);
      bf16x16 bn[4];
#pragma unroll
      for (int g = 0; g < 4; ++g) bn[g] = load_b_frag(Bp[g], UDIM, kn);
#pragma unroll
      for (int g = 0; g < 4; ++g) acc[g] = wmma_bf16(a, bf[g], acc[g]);
      a = an;
#pragma unroll
      for (int g = 0; g < 4; ++g) bf[g] = bn[g];
    }

    __syncthreads();   // all waves finished reading h_{t-1} from LDS

    // gates (i,f,g,o quarters), cell + hidden update, write h_t to LDS
    {
      int col = nblk * 16 + n;
#pragma unroll
      for (int j = 0; j < 8; ++j) {
        float iv = 1.f / (1.f + __expf(-acc[0][j]));
        float fv = 1.f / (1.f + __expf(-acc[1][j]));
        float gv = fmaxf(acc[2][j], 0.f);             // activation='relu'
        float ov = 1.f / (1.f + __expf(-acc[3][j]));
        float cv = fv * cstate[j] + iv * gv;
        cstate[j] = cv;
        float hv = ov * fmaxf(cv, 0.f);               // relu on cell
        hbuf[(j + 8 * hi) * UDIM + col] = f2bf(hv);
      }
    }
    __syncthreads();   // h_t complete

    // stream h_t to global (rows t*B+b0 .. +15 are contiguous) for next layer
    unsigned* dst = (unsigned*)(Hout + ((size_t)t * BATCH + b0) * UDIM);
    const unsigned* src = (const unsigned*)hbuf;
    for (int i = tid; i < 16 * UDIM / 2; i += 512) dst[i] = src[i];
  }
}

// ---- scatter dense result (t*B+b, NDOUTP) -> out (B,N,T,FOUT) ----
__global__ void __launch_bounds__(256) scatter_out_k(const float* __restrict__ Y,
                                                     float* __restrict__ out) {
  size_t i = (size_t)blockIdx.x * 256 + threadIdx.x;
  const size_t total = (size_t)BATCH * NPTS * TLEN * FOUT;
  if (i >= total) return;
  int f = (int)(i % FOUT);
  size_t r = i / FOUT;
  int t  = (int)(r % TLEN);  r /= TLEN;
  int nn = (int)(r % NPTS);
  int b  = (int)(r / NPTS);
  out[i] = Y[((size_t)t * BATCH + b) * NDOUTP + nn * FOUT + f];
}

extern "C" void kernel_launch(void* const* d_in, const int* in_sizes, int n_in,
                              void* d_out, int out_size, void* d_ws, size_t ws_size,
                              hipStream_t stream) {
  const float* x  = (const float*)d_in[0];
  const float* W0 = (const float*)d_in[1];
  const float* U0 = (const float*)d_in[2];
  const float* b0 = (const float*)d_in[3];
  const float* W1 = (const float*)d_in[4];
  const float* U1 = (const float*)d_in[5];
  const float* b1 = (const float*)d_in[6];
  const float* Wd = (const float*)d_in[7];
  const float* bd = (const float*)d_in[8];
  float* out = (float*)d_out;

  char* p = (char*)d_ws;
  auto carve = [&](size_t bytes) {
    char* r = p; p += (bytes + 255) & ~(size_t)255; return r;
  };
  uint16_t* XA  = (uint16_t*)carve((size_t)BT * KINP * 2);       // ~29 MB
  uint16_t* W0t = (uint16_t*)carve((size_t)GDIM * KINP * 2);
  uint16_t* U0t = (uint16_t*)carve((size_t)GDIM * UDIM * 2);
  uint16_t* W1t = (uint16_t*)carve((size_t)GDIM * UDIM * 2);
  uint16_t* U1t = (uint16_t*)carve((size_t)GDIM * UDIM * 2);
  uint16_t* Wdt = (uint16_t*)carve((size_t)NDOUTP * UDIM * 2);
  float*    ZX  = (float*)carve((size_t)BT * GDIM * 4);          // 64 MB, reused x3
  uint16_t* H0  = (uint16_t*)carve((size_t)BT * UDIM * 2);
  uint16_t* H1  = (uint16_t*)carve((size_t)BT * UDIM * 2);

  auto cdiv = [](size_t a, size_t b) { return (unsigned)((a + b - 1) / b); };

  prep_xa_k<<<cdiv((size_t)BT * KINP, 256), 256, 0, stream>>>(x, XA);
  convert_wt_k<<<cdiv((size_t)GDIM * KINP, 256), 256, 0, stream>>>(W0, W0t, KIN, GDIM, KINP, GDIM);
  convert_wt_k<<<cdiv((size_t)GDIM * UDIM, 256), 256, 0, stream>>>(U0, U0t, UDIM, GDIM, UDIM, GDIM);
  convert_wt_k<<<cdiv((size_t)GDIM * UDIM, 256), 256, 0, stream>>>(W1, W1t, UDIM, GDIM, UDIM, GDIM);
  convert_wt_k<<<cdiv((size_t)GDIM * UDIM, 256), 256, 0, stream>>>(U1, U1t, UDIM, GDIM, UDIM, GDIM);
  convert_wt_k<<<cdiv((size_t)NDOUTP * UDIM, 256), 256, 0, stream>>>(Wd, Wdt, UDIM, NDOUT, UDIM, NDOUTP);

  // layer 0: zx0 = XA @ W0^T + b0 ; recurrence
  gemm_bf16<4><<<dim3(BT / 128, GDIM / 64), 256, 0, stream>>>(
      XA, KINP, W0t, KINP, b0, GDIM, ZX, GDIM, KINP);
  lstm_rec<<<BATCH / 16, 512, 0, stream>>>(ZX, U0t, H0);

  // layer 1: zx1 = H0 @ W1^T + b1 ; recurrence
  gemm_bf16<4><<<dim3(BT / 128, GDIM / 64), 256, 0, stream>>>(
      H0, UDIM, W1t, UDIM, b1, GDIM, ZX, GDIM, UDIM);
  lstm_rec<<<BATCH / 16, 512, 0, stream>>>(ZX, U1t, H1);

  // dense head: Y = H1 @ Wd^T + bd (N padded to 640), scatter to (B,N,T,FOUT)
  gemm_bf16<4><<<dim3(BT / 128, NDOUTP / 64), 256, 0, stream>>>(
      H1, UDIM, Wdt, UDIM, bd, NDOUT, ZX, NDOUTP, UDIM);
  scatter_out_k<<<cdiv((size_t)BATCH * NPTS * TLEN * FOUT, 256), 256, 0, stream>>>(ZX, out);
}
